// ShellConv_77403900608541
// MI455X (gfx1250) — compile-verified, matrix operand order checked by reading, not run on hardware
//
#include <hip/hip_runtime.h>
#include <hip/hip_bf16.h>
#include <stdint.h>

typedef __attribute__((ext_vector_type(16))) _Float16 v16h;
typedef __attribute__((ext_vector_type(8)))  _Float16 v8h;
typedef __attribute__((ext_vector_type(8)))  float    v8f;
typedef __attribute__((address_space(3)))    _Float16 lds_f16_t;

namespace sc {
constexpr int B_ = 2, L_ = 4, N_ = 2048, M_ = 1024;
constexpr int K_ = 32, DIV_ = 4, S_ = 8;
constexpr int F1_ = 128, PF_ = 64, PF2_ = 32, OUT_ = 256, CIN_ = 192;
constexpr int R_  = B_ * L_ * M_;      // 8192 query rows (B*L*M)
constexpr int KJ_ = DIV_ * CIN_;       // 768 = GEMM reduction dim
constexpr int CNT1 = R_ * K_;          // 262144 elems/channel for bn1, bn2
constexpr int CNT3 = R_ * DIV_;        // 32768 elems/channel for bnc
}

__device__ __forceinline__ float wave_reduce_add(float v) {
#pragma unroll
  for (int m = 16; m > 0; m >>= 1) v += __shfl_xor(v, m, 32);
  return v;
}

// ---------------------------------------------------------------------------
// 1) KNN (top-32 by squared distance, stable ties) + local = q - p[idx]
//    + per-block partial sum/sumsq for bn1 (3 channels)
// ---------------------------------------------------------------------------
__global__ void __launch_bounds__(256)
k_knn_local_stats(const float* __restrict__ points,
                  const float* __restrict__ queries,
                  int* __restrict__ idx_out,
                  float* __restrict__ local_out,
                  float* __restrict__ partial1)   // [nblocks][6]
{
  using namespace sc;
  const int r    = blockIdx.x * blockDim.x + threadIdx.x;
  const int lane = threadIdx.x & 31;
  const int wave = threadIdx.x >> 5;
  const bool ok  = (r < R_);

  float s[6] = {0.f, 0.f, 0.f, 0.f, 0.f, 0.f};
  if (ok) {
    const int bl = r / M_;
    const float* __restrict__ P = points + (size_t)bl * N_ * 3;
    const float* __restrict__ q = queries + (size_t)r * 3;
    const float qx = q[0], qy = q[1], qz = q[2];

    float bd[K_];
    int   bi[K_];
#pragma unroll
    for (int k = 0; k < K_; ++k) { bd[k] = 3.0e38f; bi[k] = 0; }

    for (int n = 0; n < N_; ++n) {
      const float dx = qx - P[n * 3 + 0];
      const float dy = qy - P[n * 3 + 1];
      const float dz = qz - P[n * 3 + 2];
      const float d2 = dx * dx + dy * dy + dz * dz;
      if (d2 < bd[K_ - 1]) {
        int p = K_ - 1;
        while (p > 0 && bd[p - 1] > d2) { bd[p] = bd[p - 1]; bi[p] = bi[p - 1]; --p; }
        bd[p] = d2; bi[p] = n;
      }
    }
#pragma unroll 4
    for (int k = 0; k < K_; ++k) {
      const int n = bi[k];
      idx_out[(size_t)r * K_ + k] = n;
      const float lx = qx - P[n * 3 + 0];
      const float ly = qy - P[n * 3 + 1];
      const float lz = qz - P[n * 3 + 2];
      float* lp = local_out + ((size_t)r * K_ + k) * 3;
      lp[0] = lx; lp[1] = ly; lp[2] = lz;
      s[0] += lx;       s[1] += ly;       s[2] += lz;
      s[3] += lx * lx;  s[4] += ly * ly;  s[5] += lz * lz;
    }
  }
  __shared__ float part[8][6];
#pragma unroll
  for (int j = 0; j < 6; ++j) {
    const float w = wave_reduce_add(s[j]);
    if (lane == 0) part[wave][j] = w;
  }
  __syncthreads();
  if (threadIdx.x < 6) {
    float t = 0.f;
#pragma unroll
    for (int w = 0; w < 8; ++w) t += part[w][threadIdx.x];
    partial1[blockIdx.x * 6 + threadIdx.x] = t;
  }
}

// ---------------------------------------------------------------------------
// Finalize batchnorm: fixed-order sum over per-block partials -> a, c so that
// bn(x) = x*a + c  (biased variance, training mode; deterministic).
// partial layout: [nblocks][2*C]  (sums then sumsqs)
// ---------------------------------------------------------------------------
__global__ void __launch_bounds__(256)
k_finalize_bn(const float* __restrict__ partial, int nblocks, int C,
              const float* __restrict__ g, const float* __restrict__ beta,
              float cnt_inv, float* __restrict__ a_out, float* __restrict__ c_out)
{
  const int c = blockIdx.x * blockDim.x + threadIdx.x;
  if (c >= C) return;
  float s = 0.f, q = 0.f;
  for (int b = 0; b < nblocks; ++b) {
    s += partial[(size_t)b * (2 * C) + c];
    q += partial[(size_t)b * (2 * C) + C + c];
  }
  const float mean = s * cnt_inv;
  const float var  = q * cnt_inv - mean * mean;
  const float a = g[c] * rsqrtf(var + 1e-5f);
  a_out[c] = a;
  c_out[c] = beta[c] - mean * a;
}

// ---------------------------------------------------------------------------
// 2) Lift layer 1: h1 = relu(bn1(local) @ w1^T + b1), 3 -> 32
//    + per-block partial stats for bn2 (32 channels)
// ---------------------------------------------------------------------------
__global__ void __launch_bounds__(256)
k_lift1_stats(const float* __restrict__ local,
              const float* __restrict__ a1, const float* __restrict__ c1,
              const float* __restrict__ w1, const float* __restrict__ b1,
              float* __restrict__ h1,
              float* __restrict__ partial2)      // [nblocks][64]
{
  using namespace sc;
  const int p    = blockIdx.x * blockDim.x + threadIdx.x;
  const int lane = threadIdx.x & 31;
  const int wave = threadIdx.x >> 5;
  const bool ok  = (p < CNT1);

  __shared__ float w1s[PF2_ * 3 + PF2_];
  for (int i = threadIdx.x; i < PF2_ * 3; i += blockDim.x) w1s[i] = w1[i];
  for (int i = threadIdx.x; i < PF2_;     i += blockDim.x) w1s[PF2_ * 3 + i] = b1[i];
  __syncthreads();

  float x0 = 0.f, x1 = 0.f, x2 = 0.f;
  if (ok) {
    const float* lp = local + (size_t)p * 3;
    x0 = lp[0] * a1[0] + c1[0];
    x1 = lp[1] * a1[1] + c1[1];
    x2 = lp[2] * a1[2] + c1[2];
  }
  __shared__ float part[8][2 * PF2_];
#pragma unroll
  for (int o = 0; o < PF2_; ++o) {
    float v = w1s[o * 3 + 0] * x0 + w1s[o * 3 + 1] * x1 + w1s[o * 3 + 2] * x2
            + w1s[PF2_ * 3 + o];
    v = fmaxf(v, 0.f);
    if (ok) h1[(size_t)p * PF2_ + o] = v;
    else    v = 0.f;
    const float ws = wave_reduce_add(v);
    const float wq = wave_reduce_add(v * v);
    if (lane == 0) { part[wave][o] = ws; part[wave][PF2_ + o] = wq; }
  }
  __syncthreads();
  for (int j = threadIdx.x; j < 2 * PF2_; j += blockDim.x) {
    float t = 0.f;
#pragma unroll
    for (int w = 0; w < 8; ++w) t += part[w][j];
    partial2[(size_t)blockIdx.x * (2 * PF2_) + j] = t;
  }
}

// ---------------------------------------------------------------------------
// 3) Lift layer 2 (bn2 -> 32->64 -> relu), gather feat_prev, concat,
//    shell maxpool over S=8 neighbors -> fm[R][DIV][192]
//    + per-block partial stats for bnc (192 channels)
// ---------------------------------------------------------------------------
__global__ void __launch_bounds__(256)
k_lift2_pool_stats(const int* __restrict__ idx,
                   const float* __restrict__ h1,
                   const float* __restrict__ a2, const float* __restrict__ c2,
                   const float* __restrict__ w2, const float* __restrict__ b2,
                   const float* __restrict__ feat_prev,
                   float* __restrict__ fm,
                   float* __restrict__ partial3)  // [nblocks][384]
{
  using namespace sc;
  const int cell = blockIdx.x * blockDim.x + threadIdx.x;
  const int lane = threadIdx.x & 31;
  const int wave = threadIdx.x >> 5;
  const bool ok  = (cell < CNT3);

  __shared__ float w2s[PF_ * PF2_];
  __shared__ float b2s[PF_];
  __shared__ float a2s[PF2_], c2s[PF2_];
  for (int i = threadIdx.x; i < PF_ * PF2_; i += blockDim.x) w2s[i] = w2[i];
  if (threadIdx.x < PF_)  b2s[threadIdx.x] = b2[threadIdx.x];
  if (threadIdx.x < PF2_) { a2s[threadIdx.x] = a2[threadIdx.x]; c2s[threadIdx.x] = c2[threadIdx.x]; }
  __syncthreads();

  float mx[CIN_];
#pragma unroll
  for (int c = 0; c < CIN_; ++c) mx[c] = -3.0e38f;

  if (ok) {
    const int r  = cell >> 2;   // / DIV_
    const int d  = cell & 3;
    const int bl = r / M_;
    for (int s = 0; s < S_; ++s) {
      const size_t base = (size_t)r * K_ + d * S_ + s;
      const float* h1p = h1 + base * PF2_;
      float t[PF2_];
#pragma unroll
      for (int i = 0; i < PF2_; ++i) t[i] = h1p[i] * a2s[i] + c2s[i];
#pragma unroll 4
      for (int o = 0; o < PF_; ++o) {
        float v = b2s[o];
#pragma unroll
        for (int i = 0; i < PF2_; ++i) v += w2s[o * PF2_ + i] * t[i];
        v = fmaxf(v, 0.f);
        mx[o] = fmaxf(mx[o], v);
      }
      const int n = idx[base];
      const float* fp = feat_prev + ((size_t)bl * N_ + n) * F1_;
#pragma unroll 8
      for (int c = 0; c < F1_; ++c) mx[PF_ + c] = fmaxf(mx[PF_ + c], fp[c]);
    }
  }

  __shared__ float part[8][2 * CIN_];
  float* fout = fm + (size_t)cell * CIN_;
  for (int c = 0; c < CIN_; ++c) {
    float v = ok ? mx[c] : 0.f;
    if (ok) fout[c] = v;
    const float ws = wave_reduce_add(v);
    const float wq = wave_reduce_add(v * v);
    if (lane == 0) { part[wave][c] = ws; part[wave][CIN_ + c] = wq; }
  }
  __syncthreads();
  for (int j = threadIdx.x; j < 2 * CIN_; j += blockDim.x) {
    float t = 0.f;
#pragma unroll
    for (int w = 0; w < 8; ++w) t += part[w][j];
    partial3[(size_t)blockIdx.x * (2 * CIN_) + j] = t;
  }
}

// ---------------------------------------------------------------------------
// 4) Pack GEMM operands to f16:
//    Xh[r][j=d*192+c] = bnc(fm[r][d][c]),  Wh[o][j=d*192+c] = conv_w[o][c][d]
// ---------------------------------------------------------------------------
__global__ void __launch_bounds__(256)
k_pack_x(const float* __restrict__ fm,
         const float* __restrict__ a3, const float* __restrict__ c3,
         _Float16* __restrict__ Xh)
{
  using namespace sc;
  const int e = blockIdx.x * blockDim.x + threadIdx.x;
  if (e >= R_ * KJ_) return;
  const int r = e / KJ_;
  const int j = e - r * KJ_;
  const int d = j / CIN_;
  const int c = j - d * CIN_;
  const float v = fm[((size_t)r * DIV_ + d) * CIN_ + c] * a3[c] + c3[c];
  Xh[e] = (_Float16)v;
}

__global__ void __launch_bounds__(256)
k_pack_w(const float* __restrict__ conv_w, _Float16* __restrict__ Wh)
{
  using namespace sc;
  const int e = blockIdx.x * blockDim.x + threadIdx.x;
  if (e >= OUT_ * KJ_) return;
  const int o = e / KJ_;
  const int j = e - o * KJ_;
  const int d = j / CIN_;
  const int c = j - d * CIN_;
  Wh[e] = (_Float16)conv_w[((size_t)o * CIN_ + c) * DIV_ + d];
}

// ---------------------------------------------------------------------------
// 5) WMMA GEMM: out[8192 x 256] = Xh[8192 x 768] * Wh[256 x 768]^T + conv_b
//
//    Block = 256 threads (8 waves) -> 64x64 output tile.
//    Double-buffered LDS staging via GLOBAL_LOAD_ASYNC_TO_LDS_B128 (ASYNCcnt),
//    waited with s_wait_asynccnt (async loads complete in order, so after
//    issuing the next chunk's 2 loads, "cnt<=2" == current chunk landed).
//    LDS rows padded to 80B so 16-lane ds_load_b128 column reads are
//    conflict-free (20 dwords/row, gcd(20,64)=4, cycle 16).
//    Each wave: one 16x16 A tile (lane m=lane&15, hi=lane>>4; halfs 0-7 ->
//    K in [hi*8,hi*8+8), halfs 8-15 -> K in [16+hi*8,...)) against two B
//    tiles (lane = N column of W^T, same K striping) -> 2 WMMA per k-step.
// ---------------------------------------------------------------------------
__global__ void __launch_bounds__(256)
k_conv_wmma(const _Float16* __restrict__ Xh,
            const _Float16* __restrict__ Wh,
            const float* __restrict__ conv_b,
            float* __restrict__ out)
{
  using namespace sc;
  __shared__ __align__(16) _Float16 Ash[2][64][40];  // 2 x 64 rows x 80B
  __shared__ __align__(16) _Float16 Bsh[2][64][40];

  const int tid  = threadIdx.x;
  const int lane = tid & 31;
  const int wave = tid >> 5;
  const int m0   = blockIdx.x * 64;
  const int n0   = blockIdx.y * 64;

  // async producer addressing: thread -> (row 0..63, 16B quarter 0..3)
  const int trow = tid >> 2;
  const int tq   = tid & 3;
  const _Float16* gA = Xh + (size_t)(m0 + trow) * KJ_ + tq * 8;
  const _Float16* gB = Wh + (size_t)(n0 + trow) * KJ_ + tq * 8;
  const uint32_t ldsA0 =
      (uint32_t)(size_t)(lds_f16_t*)&Ash[0][trow][tq * 8];
  const uint32_t ldsB0 =
      (uint32_t)(size_t)(lds_f16_t*)&Bsh[0][trow][tq * 8];
  constexpr uint32_t BUFB = 64u * 40u * (uint32_t)sizeof(_Float16); // 5120B

  // consumer addressing
  const int mi    = wave & 3;          // m sub-tile 0..3
  const int nt0   = (wave >> 2) * 2;   // first of two n sub-tiles
  const int row16 = lane & 15;
  const int hi    = lane >> 4;

  v8f acc0 = {}, acc1 = {};

  // prologue: stage k-chunk 0 into buffer 0 (16B per lane per matrix)
  asm volatile("global_load_async_to_lds_b128 %0, %1, off"
               :: "v"(ldsA0), "v"(gA) : "memory");
  asm volatile("global_load_async_to_lds_b128 %0, %1, off"
               :: "v"(ldsB0), "v"(gB) : "memory");

  int buf = 0;
  for (int kk = 0; kk < KJ_ / 32; ++kk) {
    if (kk + 1 < KJ_ / 32) {
      const uint32_t nb = (uint32_t)(buf ^ 1) * BUFB;
      asm volatile("global_load_async_to_lds_b128 %0, %1, off"
                   :: "v"(ldsA0 + nb), "v"(gA + (kk + 1) * 32) : "memory");
      asm volatile("global_load_async_to_lds_b128 %0, %1, off"
                   :: "v"(ldsB0 + nb), "v"(gB + (kk + 1) * 32) : "memory");
      asm volatile("s_wait_asynccnt 0x2" ::: "memory");
    } else {
      asm volatile("s_wait_asynccnt 0x0" ::: "memory");
    }
    __syncthreads();   // everyone's current chunk is resident

    const v8h alo  = *(const v8h*)&Ash[buf][mi * 16 + row16][hi * 8];
    const v8h ahi  = *(const v8h*)&Ash[buf][mi * 16 + row16][16 + hi * 8];
    const v8h b0lo = *(const v8h*)&Bsh[buf][nt0 * 16 + row16][hi * 8];
    const v8h b0hi = *(const v8h*)&Bsh[buf][nt0 * 16 + row16][16 + hi * 8];
    const v8h b1lo = *(const v8h*)&Bsh[buf][(nt0 + 1) * 16 + row16][hi * 8];
    const v8h b1hi = *(const v8h*)&Bsh[buf][(nt0 + 1) * 16 + row16][16 + hi * 8];
    v16h a, b0, b1;
#pragma unroll
    for (int i = 0; i < 8; ++i) {
      a[i]  = alo[i];  a[i + 8]  = ahi[i];
      b0[i] = b0lo[i]; b0[i + 8] = b0hi[i];
      b1[i] = b1lo[i]; b1[i + 8] = b1hi[i];
    }
    acc0 = __builtin_amdgcn_wmma_f32_16x16x32_f16(
        false, a, false, b0, (short)0, acc0, false, false);
    acc1 = __builtin_amdgcn_wmma_f32_16x16x32_f16(
        false, a, false, b1, (short)0, acc1, false, false);
    __syncthreads();   // done reading buf before it is overwritten
    buf ^= 1;
  }

  const float bias0 = conv_b[n0 + nt0 * 16 + row16];
  const float bias1 = conv_b[n0 + (nt0 + 1) * 16 + row16];
#pragma unroll
  for (int v = 0; v < 8; ++v) {  // C layout: VGPR v -> M = v + hi*8, N = row16
    const size_t rrow = (size_t)(m0 + mi * 16 + v + hi * 8) * OUT_;
    out[rrow + n0 + nt0 * 16 + row16]       = acc0[v] + bias0;
    out[rrow + n0 + (nt0 + 1) * 16 + row16] = acc1[v] + bias1;
  }
}

// ---------------------------------------------------------------------------
extern "C" void kernel_launch(void* const* d_in, const int* in_sizes, int n_in,
                              void* d_out, int out_size, void* d_ws, size_t ws_size,
                              hipStream_t stream)
{
  using namespace sc;
  (void)in_sizes; (void)n_in; (void)out_size; (void)ws_size;
  const float* points    = (const float*)d_in[0];
  const float* queries   = (const float*)d_in[1];
  const float* feat_prev = (const float*)d_in[2];
  const float* bn1_g  = (const float*)d_in[3];
  const float* bn1_b  = (const float*)d_in[4];
  const float* w1     = (const float*)d_in[5];
  const float* b1     = (const float*)d_in[6];
  const float* bn2_g  = (const float*)d_in[7];
  const float* bn2_b  = (const float*)d_in[8];
  const float* w2     = (const float*)d_in[9];
  const float* b2     = (const float*)d_in[10];
  const float* bnc_g  = (const float*)d_in[11];
  const float* bnc_b  = (const float*)d_in[12];
  const float* conv_w = (const float*)d_in[13];
  const float* conv_b = (const float*)d_in[14];
  float* out = (float*)d_out;

  char* ws = (char*)d_ws;
  size_t off = 0;
  auto alloc = [&](size_t bytes) -> char* {
    char* p = ws + off;
    off += (bytes + 255) & ~(size_t)255;
    return p;
  };
  int*      ws_idx   = (int*)     alloc((size_t)R_ * K_ * sizeof(int));
  float*    ws_local = (float*)   alloc((size_t)R_ * K_ * 3 * sizeof(float));
  float*    ws_h1    = (float*)   alloc((size_t)R_ * K_ * PF2_ * sizeof(float));
  float*    ws_fm    = (float*)   alloc((size_t)R_ * DIV_ * CIN_ * sizeof(float));
  _Float16* ws_Xh    = (_Float16*)alloc((size_t)R_ * KJ_ * sizeof(_Float16));
  _Float16* ws_Wh    = (_Float16*)alloc((size_t)OUT_ * KJ_ * sizeof(_Float16));
  float*    ws_p1    = (float*)   alloc((size_t)(R_ / 256) * 6 * sizeof(float));
  float*    ws_p2    = (float*)   alloc((size_t)(CNT1 / 256) * 64 * sizeof(float));
  float*    ws_p3    = (float*)   alloc((size_t)(CNT3 / 256) * 384 * sizeof(float));
  float*    ws_bn    = (float*)   alloc((size_t)512 * sizeof(float));
  float* A1 = ws_bn + 0;   float* C1 = ws_bn + 4;
  float* A2 = ws_bn + 8;   float* C2 = ws_bn + 40;
  float* A3 = ws_bn + 72;  float* C3 = ws_bn + 264;

  // 1) KNN + local offsets + bn1 partial stats
  k_knn_local_stats<<<dim3(R_ / 256), dim3(256), 0, stream>>>(
      points, queries, ws_idx, ws_local, ws_p1);
  k_finalize_bn<<<dim3(1), dim3(256), 0, stream>>>(
      ws_p1, R_ / 256, 3, bn1_g, bn1_b, 1.0f / (float)CNT1, A1, C1);

  // 2) lift layer 1 + bn2 partial stats
  k_lift1_stats<<<dim3(CNT1 / 256), dim3(256), 0, stream>>>(
      ws_local, A1, C1, w1, b1, ws_h1, ws_p2);
  k_finalize_bn<<<dim3(1), dim3(256), 0, stream>>>(
      ws_p2, CNT1 / 256, PF2_, bn2_g, bn2_b, 1.0f / (float)CNT1, A2, C2);

  // 3) lift layer 2 + gather prev feats + shell maxpool + bnc partial stats
  k_lift2_pool_stats<<<dim3(CNT3 / 256), dim3(256), 0, stream>>>(
      ws_idx, ws_h1, A2, C2, w2, b2, feat_prev, ws_fm, ws_p3);
  k_finalize_bn<<<dim3(1), dim3(256), 0, stream>>>(
      ws_p3, CNT3 / 256, CIN_, bnc_g, bnc_b, 1.0f / (float)CNT3, A3, C3);

  // 4) pack normalized activations and transposed weights to f16
  k_pack_x<<<dim3((R_ * KJ_) / 256), dim3(256), 0, stream>>>(ws_fm, A3, C3, ws_Xh);
  k_pack_w<<<dim3((OUT_ * KJ_) / 256), dim3(256), 0, stream>>>(conv_w, ws_Wh);

  // 5) WMMA GEMM + bias, async-LDS double-buffered, 64x64 tile per block
  k_conv_wmma<<<dim3(R_ / 64, OUT_ / 64), dim3(256), 0, stream>>>(
      ws_Xh, ws_Wh, conv_b, out);
}
